// LunaBlock_85341000171970
// MI455X (gfx1250) — matrix operand order repaired and checked by reading, not compiled, not run on hardware
//
#include <hip/hip_runtime.h>

typedef __attribute__((ext_vector_type(16))) _Float16 v16h;
typedef __attribute__((ext_vector_type(8)))  _Float16 v8h;
typedef __attribute__((ext_vector_type(8)))  float    v8f;

namespace {

constexpr int kB   = 8;
constexpr int kHW  = 4096;
constexpr int kK   = 256;
constexpr int kHID = 1024;
constexpr int kAUX = 512;
constexpr int kQK  = 512;
constexpr int kNH  = 8;
constexpr float kSCALE = 0.08838834764831845f;  // 1/sqrt(HEAD=128)
constexpr float kEPS = 1e-5f;

// ---------------- f32 -> f16 bulk convert (32 B in / 16 B out per lane) ----

__global__ __launch_bounds__(256) void cvt_f32_f16(
    const float* __restrict__ x, _Float16* __restrict__ y, long long n) {
  const long long i = ((long long)blockIdx.x * 256 + threadIdx.x) * 8;
  if (i >= n) return;
  const float4 a = *(const float4*)(x + i);
  const float4 b = *(const float4*)(x + i + 4);
  v8h o;
  o[0] = (_Float16)a.x; o[1] = (_Float16)a.y;
  o[2] = (_Float16)a.z; o[3] = (_Float16)a.w;
  o[4] = (_Float16)b.x; o[5] = (_Float16)b.y;
  o[6] = (_Float16)b.z; o[7] = (_Float16)b.w;
  *(v8h*)(y + i) = o;
}

// ---------------- WMMA fragment loaders (ISA 7.12.2 layouts, wave32) -------

// A fragment 16x32 (MxK), f16 source. Lane: row = lane&15, kb = (lane>>4)*8,
// elems = A[row][kb..kb+7] then A[row][16+kb..16+kb+7]  (two 16 B loads).
__device__ __forceinline__ v16h load_a_frag(const _Float16* __restrict__ base,
                                            int lda, int lane) {
  const int row = lane & 15;
  const int kb  = (lane >> 4) * 8;
  const _Float16* p = base + (long long)row * lda;
  const v8h lo = *(const v8h*)(p + kb);
  const v8h hi = *(const v8h*)(p + 16 + kb);
  v16h a;
#pragma unroll
  for (int j = 0; j < 8; ++j) { a[j] = lo[j]; a[8 + j] = hi[j]; }
  return a;
}

// A fragment from f32 source (used for softmax probabilities), cvt in-loop.
__device__ __forceinline__ v16h load_a_frag(const float* __restrict__ base,
                                            int lda, int lane) {
  const int row = lane & 15;
  const int kb  = (lane >> 4) * 8;
  const float* p = base + (long long)row * lda;
  v16h a;
#pragma unroll
  for (int j = 0; j < 8; ++j) a[j] = (_Float16)p[kb + j];
#pragma unroll
  for (int j = 0; j < 8; ++j) a[8 + j] = (_Float16)p[16 + kb + j];
  return a;
}

// B fragment (32x16 KxN) from a K-major f16 matrix Bt[N][Kd] (B = Bt^T):
// weights W[N][Kd] or transposed V (d, token). Lane: col = lane&15, kb = (lane>>4)*8,
// elems = K kb..kb+7 then 16+kb..16+kb+7 (two 16 B loads).
__device__ __forceinline__ v16h load_b_frag_t(const _Float16* __restrict__ base,
                                              int ldw, int lane) {
  const int col = lane & 15;
  const int kb  = (lane >> 4) * 8;
  const _Float16* p = base + (long long)col * ldw + kb;
  const v8h lo = *(const v8h*)(p);
  const v8h hi = *(const v8h*)(p + 16);
  v16h b;
#pragma unroll
  for (int j = 0; j < 8; ++j) { b[j] = lo[j]; b[8 + j] = hi[j]; }
  return b;
}

// ---------------- generic wave-tiled WMMA GEMM ------------------------------
// Block = 256 threads = 8 waves (BMW x BNW); wave strip = 16 x 64 (4 accums).
// C = scale * A @ Bt^T (+bias[col]) (+res[r,c]); A is f16 or f32 (TA), Bt f16
// K-major. Outputs: CT=false -> Cf (f32) and/or Ch (f16), row-major [M,ldc].
//           CT=true  -> Ch only, TRANSPOSED [N rows, ldc cols]: Ch[c*ldc + r]
//                       (8 contiguous rows per lane -> one 16 B store/tile).
// Batched over blockIdx.z: zb = z/nh, zh = z%nh (per-(batch,head) strides).
template <int BMW, int BNW, bool CT, typename TA>
__global__ __launch_bounds__(256) void wmma_gemm(
    const TA* __restrict__ A, int lda, long long sAb, long long sAh,
    const _Float16* __restrict__ Bm, int ldb, long long sBb, long long sBh,
    const float* __restrict__ bias, const float* __restrict__ res,
    float* __restrict__ Cf, _Float16* __restrict__ Ch,
    int ldc, long long sCb, long long sCh,
    int Kd, int nh, float scale) {
  const int zb   = blockIdx.z / nh;
  const int zh   = blockIdx.z % nh;
  const int w    = threadIdx.x >> 5;
  const int lane = threadIdx.x & 31;
  const int wm   = w / BNW;
  const int wn   = w % BNW;
  const int r0   = blockIdx.x * (16 * BMW) + wm * 16;
  const int c0   = blockIdx.y * (64 * BNW) + wn * 64;

  const TA* pA = A + (long long)zb * sAb + (long long)zh * sAh +
                 (long long)r0 * lda;
  const _Float16* pB = Bm + (long long)zb * sBb + (long long)zh * sBh;

  v8f acc[4];
#pragma unroll
  for (int t = 0; t < 4; ++t)
#pragma unroll
    for (int j = 0; j < 8; ++j) acc[t][j] = 0.0f;

  for (int kk = 0; kk < Kd; kk += 32) {
    if (kk + 128 < Kd)  // prefetch A panel ahead (global_prefetch_b8)
      __builtin_prefetch((const void*)(pA + kk + 128), 0, 1);
    const v16h a = load_a_frag(pA + kk, lda, lane);
#pragma unroll
    for (int t = 0; t < 4; ++t) {
      const v16h b =
          load_b_frag_t(pB + (long long)(c0 + 16 * t) * ldb + kk, ldb, lane);
      acc[t] = __builtin_amdgcn_wmma_f32_16x16x32_f16(
          false, a, false, b, (short)0, acc[t], false, false);
    }
  }

  // C/D layout: col = lane&15, rows = 8*(lane>>4) + vgpr index.
  const int coll = lane & 15;
  const int rb   = (lane >> 4) * 8;
  const long long zoff = (long long)zb * sCb + (long long)zh * sCh;
  if constexpr (CT) {
    _Float16* pCh = Ch + zoff;
#pragma unroll
    for (int t = 0; t < 4; ++t) {
      const int col  = c0 + 16 * t + coll;
      const float bv = bias ? bias[col] : 0.0f;
      v8h o;
#pragma unroll
      for (int j = 0; j < 8; ++j) o[j] = (_Float16)(acc[t][j] * scale + bv);
      *(v8h*)(pCh + (long long)col * ldc + (r0 + rb)) = o;
    }
  } else {
    float* pCf = Cf ? Cf + zoff : nullptr;
    _Float16* pCh = Ch ? Ch + zoff : nullptr;
    const float* pR = res ? res + zoff : nullptr;
#pragma unroll
    for (int t = 0; t < 4; ++t) {
      const int col  = c0 + 16 * t + coll;
      const float bv = bias ? bias[col] : 0.0f;
#pragma unroll
      for (int j = 0; j < 8; ++j) {
        const long long idx = (long long)(r0 + rb + j) * ldc + col;
        float v = acc[t][j] * scale + bv;
        if (pR) v += pR[idx];
        if (pCf) pCf[idx] = v;
        if (pCh) pCh[idx] = (_Float16)v;
      }
    }
  }
}

// ---------------- block reductions ------------------------------------------

__device__ __forceinline__ float block_sum(float v, float* sm) {
  const int t = threadIdx.x;
  sm[t] = v;
  __syncthreads();
#pragma unroll
  for (int s = 128; s > 0; s >>= 1) {
    if (t < s) sm[t] += sm[t + s];
    __syncthreads();
  }
  const float r = sm[0];
  __syncthreads();
  return r;
}

__device__ __forceinline__ float block_max(float v, float* sm) {
  const int t = threadIdx.x;
  sm[t] = v;
  __syncthreads();
#pragma unroll
  for (int s = 128; s > 0; s >>= 1) {
    if (t < s) sm[t] = fmaxf(sm[t], sm[t + s]);
    __syncthreads();
  }
  const float r = sm[0];
  __syncthreads();
  return r;
}

// ---------------- in-place row softmax --------------------------------------

template <int L>
__global__ __launch_bounds__(256) void softmax_rows(float* __restrict__ S) {
  constexpr int VPT = L / 256;
  __shared__ float sm[256];
  float* p = S + (long long)blockIdx.x * L;
  const int t = threadIdx.x;
  float v[VPT];
  float m = -3.4e38f;
#pragma unroll
  for (int i = 0; i < VPT; ++i) {
    v[i] = p[t + i * 256];
    m = fmaxf(m, v[i]);
  }
  m = block_max(m, sm);
  float s = 0.0f;
#pragma unroll
  for (int i = 0; i < VPT; ++i) {
    v[i] = __expf(v[i] - m);
    s += v[i];
  }
  s = block_sum(s, sm);
  const float inv = 1.0f / s;
#pragma unroll
  for (int i = 0; i < VPT; ++i) p[t + i * 256] = v[i] * inv;
}

// ---------------- residual + LayerNorm (one block per row) ------------------

template <int C>
__global__ __launch_bounds__(256) void layernorm_rows(
    const float* __restrict__ x, const float* __restrict__ res,
    const float* __restrict__ g, const float* __restrict__ bb,
    float* __restrict__ y) {
  constexpr int VPT = C / 256;
  __shared__ float sm[256];
  const long long row = blockIdx.x;
  const float* px = x + row * C;
  const float* pr = res ? res + row * C : nullptr;
  const int t = threadIdx.x;
  float v[VPT];
  float s = 0.0f;
#pragma unroll
  for (int i = 0; i < VPT; ++i) {
    v[i] = px[t + i * 256];
    if (pr) v[i] += pr[t + i * 256];
    s += v[i];
  }
  const float mean = block_sum(s, sm) * (1.0f / C);
  float s2 = 0.0f;
#pragma unroll
  for (int i = 0; i < VPT; ++i) {
    const float d = v[i] - mean;
    s2 += d * d;
  }
  const float var  = block_sum(s2, sm) * (1.0f / C);
  const float rstd = rsqrtf(var + kEPS);
#pragma unroll
  for (int i = 0; i < VPT; ++i) {
    const int col = t + i * 256;
    y[row * C + col] = (v[i] - mean) * rstd * g[col] + bb[col];
  }
}

}  // namespace

// ---------------- orchestration ---------------------------------------------

extern "C" void kernel_launch(void* const* d_in, const int* in_sizes, int n_in,
                              void* d_out, int out_size, void* d_ws,
                              size_t ws_size, hipStream_t stream) {
  (void)in_sizes; (void)n_in; (void)out_size; (void)ws_size;
  const float* hidden = (const float*)d_in[0];   // (B,HW,HID)
  const float* aux    = (const float*)d_in[1];   // (B,K,AUX)
  const float* wq1 = (const float*)d_in[2];   const float* bq1 = (const float*)d_in[3];
  const float* wk1 = (const float*)d_in[4];   const float* bk1 = (const float*)d_in[5];
  const float* wv1 = (const float*)d_in[6];   const float* bv1 = (const float*)d_in[7];
  const float* wo1 = (const float*)d_in[8];   const float* bo1 = (const float*)d_in[9];
  const float* wq2 = (const float*)d_in[10];  const float* bq2 = (const float*)d_in[11];
  const float* wk2 = (const float*)d_in[12];  const float* bk2 = (const float*)d_in[13];
  const float* wv2 = (const float*)d_in[14];  const float* bv2 = (const float*)d_in[15];
  const float* wo2 = (const float*)d_in[16];  const float* bo2 = (const float*)d_in[17];
  const float* g_aux = (const float*)d_in[18]; const float* b_aux = (const float*)d_in[19];
  const float* g_h   = (const float*)d_in[20]; const float* b_h   = (const float*)d_in[21];

  // d_out = [out | aux_out | attn2 | attn1] (reference return order), f32.
  float* out     = (float*)d_out;                                   // B*HW*HID
  float* aux_out = out + (size_t)kB * kHW * kHID;                   // B*K*AUX
  float* attn2   = aux_out + (size_t)kB * kK * kAUX;                // B*NH*HW*K
  float* attn1   = attn2 + (size_t)kB * kNH * kHW * kK;             // B*NH*K*HW

  // ---- workspace layout (~192 MB, phase-aliased) ---------------------------
  _Float16* h16   = (_Float16*)d_ws;                 // 33,554,432 (persistent)
  _Float16* aux16 = h16 + (size_t)33554432;          //  1,048,576 (persistent)
  _Float16* wq1h  = aux16 + (size_t)1048576;         //    262,144
  _Float16* wk1h  = wq1h + (size_t)262144;           //    524,288
  _Float16* wv1h  = wk1h + (size_t)524288;           //  1,048,576
  _Float16* wo1h  = wv1h + (size_t)1048576;          //    524,288
  _Float16* wq2h  = wo1h + (size_t)524288;           //    524,288
  _Float16* wk2h  = wq2h + (size_t)524288;           //    262,144
  _Float16* wv2h  = wk2h + (size_t)262144;           //    524,288
  _Float16* wo2h  = wv2h + (size_t)524288;           //  1,048,576
  _Float16* S2    = wo2h + (size_t)1048576;          // 33,554,432: v1t -> ctx2h
  _Float16* S3    = S2 + (size_t)33554432;           // 16,777,216: k1h -> q2h
  _Float16* S4    = S3 + (size_t)16777216;           //  1,048,576: q1h -> k2h
  _Float16* S5    = S4 + (size_t)1048576;            //  2,097,152: ctx1h -> v2t
  float*    out1f = (float*)(S5 + (size_t)2097152);  //  1,048,576 f32
  _Float16* out1h = (_Float16*)(out1f + (size_t)1048576);  // 1,048,576 f16

  _Float16* v1t = S2;   _Float16* ctx2h = S2;   // v1t: (HID, B*HW) transposed
  _Float16* k1h = S3;   _Float16* q2h   = S3;
  _Float16* q1h = S4;   _Float16* k2h   = S4;
  _Float16* ctx1h = S5; _Float16* v2t   = S5;   // v2t: (HID, B*K) transposed

  const int MT = kB * kHW;  // 32768 hidden tokens
  const int MA = kB * kK;   // 2048 aux tokens
  const dim3 blk(256);
  auto cvt = [&](const float* src, _Float16* dst, long long n) {
    cvt_f32_f16<<<dim3((unsigned)(n / 2048)), blk, 0, stream>>>(src, dst, n);
  };

  // ---- one-time f32 -> f16 conversion --------------------------------------
  cvt(hidden, h16, (long long)MT * kHID);
  cvt(aux, aux16, (long long)MA * kAUX);
  cvt(wq1, wq1h, (long long)kQK * kAUX);
  cvt(wk1, wk1h, (long long)kQK * kHID);
  cvt(wv1, wv1h, (long long)kHID * kHID);
  cvt(wo1, wo1h, (long long)kAUX * kHID);
  cvt(wq2, wq2h, (long long)kQK * kHID);
  cvt(wk2, wk2h, (long long)kQK * kAUX);
  cvt(wv2, wv2h, (long long)kHID * kAUX);
  cvt(wo2, wo2h, (long long)kHID * kHID);

  // ---- pack path -----------------------------------------------------------
  // q1 = aux @ wq1^T + bq1 -> f16
  wmma_gemm<2, 4, false><<<dim3(MA / 32, kQK / 256, 1), blk, 0, stream>>>(
      aux16, kAUX, 0, 0, wq1h, kAUX, 0, 0, bq1, nullptr,
      (float*)nullptr, q1h, kQK, 0, 0, kAUX, 1, 1.0f);
  // k1 = hidden @ wk1^T + bk1 -> f16
  wmma_gemm<2, 4, false><<<dim3(MT / 32, kQK / 256, 1), blk, 0, stream>>>(
      h16, kHID, 0, 0, wk1h, kHID, 0, 0, bk1, nullptr,
      (float*)nullptr, k1h, kQK, 0, 0, kHID, 1, 1.0f);
  // v1 = hidden @ wv1^T + bv1 -> f16 TRANSPOSED (HID, B*HW)
  wmma_gemm<2, 4, true><<<dim3(MT / 32, kHID / 256, 1), blk, 0, stream>>>(
      h16, kHID, 0, 0, wv1h, kHID, 0, 0, bv1, nullptr,
      (float*)nullptr, v1t, MT, 0, 0, kHID, 1, 1.0f);

  // attn1 scores: per (b,h) 256x4096, Kd=64, scaled -> f32 (output region)
  wmma_gemm<2, 4, false><<<dim3(kK / 32, kHW / 256, kB * kNH), blk, 0, stream>>>(
      q1h, kQK, (long long)kK * kQK, 64,
      k1h, kQK, (long long)kHW * kQK, 64,
      nullptr, nullptr,
      attn1, (_Float16*)nullptr, kHW, (long long)kNH * kK * kHW,
      (long long)kK * kHW, 64, kNH, kSCALE);
  softmax_rows<kHW><<<dim3(kB * kNH * kK), blk, 0, stream>>>(attn1);
  // ctx1[:, h*128:] = attn1 @ v1_h : per (b,h) 256x128, Kd=4096 -> f16
  // B operand = v1t rows d = h*128+c (K-major over tokens of batch b).
  wmma_gemm<4, 2, false><<<dim3(kK / 64, 1, kB * kNH), blk, 0, stream>>>(
      attn1, kHW, (long long)kNH * kK * kHW, (long long)kK * kHW,
      v1t, MT, (long long)kHW, (long long)128 * MT,
      nullptr, nullptr,
      (float*)nullptr, ctx1h, kHID, (long long)kK * kHID, 128,
      kHW, kNH, 1.0f);
  // out1 = ctx1 @ wo1^T + bo1 -> f32 (for LN residual) + f16 (feeds k2/v2)
  wmma_gemm<2, 4, false><<<dim3(MA / 32, kAUX / 256, 1), blk, 0, stream>>>(
      ctx1h, kHID, 0, 0, wo1h, kHID, 0, 0, bo1, nullptr,
      out1f, out1h, kAUX, 0, 0, kHID, 1, 1.0f);
  // aux_out = LN(aux + out1)
  layernorm_rows<kAUX><<<dim3(MA), blk, 0, stream>>>(out1f, aux, g_aux, b_aux,
                                                     aux_out);

  // ---- unpack path ---------------------------------------------------------
  // q2 = hidden @ wq2^T + bq2 -> f16
  wmma_gemm<2, 4, false><<<dim3(MT / 32, kQK / 256, 1), blk, 0, stream>>>(
      h16, kHID, 0, 0, wq2h, kHID, 0, 0, bq2, nullptr,
      (float*)nullptr, q2h, kQK, 0, 0, kHID, 1, 1.0f);
  // k2 = out1 @ wk2^T + bk2 -> f16
  wmma_gemm<2, 4, false><<<dim3(MA / 32, kQK / 256, 1), blk, 0, stream>>>(
      out1h, kAUX, 0, 0, wk2h, kAUX, 0, 0, bk2, nullptr,
      (float*)nullptr, k2h, kQK, 0, 0, kAUX, 1, 1.0f);
  // v2 = out1 @ wv2^T + bv2 -> f16 TRANSPOSED (HID, B*K)
  wmma_gemm<2, 4, true><<<dim3(MA / 32, kHID / 256, 1), blk, 0, stream>>>(
      out1h, kAUX, 0, 0, wv2h, kAUX, 0, 0, bv2, nullptr,
      (float*)nullptr, v2t, MA, 0, 0, kAUX, 1, 1.0f);

  // attn2 scores: per (b,h) 4096x256, Kd=64, scaled -> f32 (output region)
  wmma_gemm<2, 4, false><<<dim3(kHW / 32, kK / 256, kB * kNH), blk, 0, stream>>>(
      q2h, kQK, (long long)kHW * kQK, 64,
      k2h, kQK, (long long)kK * kQK, 64,
      nullptr, nullptr,
      attn2, (_Float16*)nullptr, kK, (long long)kNH * kHW * kK,
      (long long)kHW * kK, 64, kNH, kSCALE);
  softmax_rows<kK><<<dim3(kB * kNH * kHW), blk, 0, stream>>>(attn2);
  // ctx2[:, h*128:] = attn2 @ v2_h : per (b,h) 4096x128, Kd=256 -> f16
  wmma_gemm<4, 2, false><<<dim3(kHW / 64, 1, kB * kNH), blk, 0, stream>>>(
      attn2, kK, (long long)kNH * kHW * kK, (long long)kHW * kK,
      v2t, MA, (long long)kK, (long long)128 * MA,
      nullptr, nullptr,
      (float*)nullptr, ctx2h, kHID, (long long)kHW * kHID, 128,
      kK, kNH, 1.0f);
  // out = ctx2 @ wo2^T + bo2 + hidden (residual fused) -> f32, LN in place.
  wmma_gemm<2, 4, false><<<dim3(MT / 32, kHID / 256, 1), blk, 0, stream>>>(
      ctx2h, kHID, 0, 0, wo2h, kHID, 0, 0, bo2, hidden,
      out, (_Float16*)nullptr, kHID, 0, 0, kHID, 1, 1.0f);
  layernorm_rows<kHID><<<dim3(MT), blk, 0, stream>>>(out, nullptr, g_h, b_h,
                                                     out);
}
